// MultiHeadedAttention_1348619731116
// MI455X (gfx1250) — matrix operand order repaired and checked by reading, not compiled, not run on hardware
//
#include <hip/hip_runtime.h>
#include <stdint.h>

#define S_LEN 1024
#define DMODEL 1024
#define NH 16
#define DH 64
#define BATCH 8

typedef __bf16 v16bf __attribute__((ext_vector_type(16)));
typedef __bf16 v2bf  __attribute__((ext_vector_type(2)));
typedef float  v8f   __attribute__((ext_vector_type(8)));
typedef float  v2f   __attribute__((ext_vector_type(2)));
typedef unsigned int v4u __attribute__((ext_vector_type(4)));
typedef int v8i __attribute__((ext_vector_type(8)));
typedef int v4i __attribute__((ext_vector_type(4)));

union FragU {
  uint32_t u[8];
  uint4    q[2];
  v16bf    v;
};

// hardware bf16 conversion (v_cvt_pk_bf16_f32 / v_cvt_bf16_f32)
__device__ __forceinline__ uint32_t pack2bf(float lo, float hi) {
  v2f f; f.x = lo; f.y = hi;
  v2bf b = __builtin_convertvector(f, v2bf);
  return __builtin_bit_cast(uint32_t, b);
}
__device__ __forceinline__ uint16_t f2bf(float f) {
  __bf16 b = (__bf16)f;
  return __builtin_bit_cast(uint16_t, b);
}
// swap data between lanes n and n^16 (SWAPX16: xor=0x10, and=0x1f)
__device__ __forceinline__ float swz16(float x) {
  int i = __builtin_amdgcn_ds_swizzle(__builtin_bit_cast(int, x), 0x401f);
  return __builtin_bit_cast(float, i);
}
// A-fragment (16x32 bf16) from an LDS row: VGPR0-3 = K kbase..kbase+7 (16B),
// VGPR4-7 = K 16+kbase..16+kbase+7 (16B at +32 bytes)
__device__ __forceinline__ v16bf lds_fragA(const uint16_t* rowp, int byteoff) {
  FragU f;
  f.q[0] = *(const uint4*)((const char*)rowp + byteoff);
  f.q[1] = *(const uint4*)((const char*)rowp + byteoff + 32);
  return f.v;
}

#define WMMA_BF16(a, b, c) \
  __builtin_amdgcn_wmma_f32_16x16x32_bf16(false, (a), false, (b), (short)0, (c), false, false)

// ---------------------------------------------------------------------------
// Tensor Data Mover: 2D tile load (bf16 elements) global -> LDS with row pad.
// Descriptor bitfields per CDNA5 ISA 8.3-8.6 (D# group0 128b + group1 256b).
// pad codes: interval 0=2dw,1=4,2=8,3=16,4=32...; amount 0=1dw,1=2,2=3,3=4...
// ---------------------------------------------------------------------------
__device__ __forceinline__ void tdm_load_2d(uint32_t lds_off, const void* gptr,
                                            uint32_t tile_w /*elems*/,
                                            uint32_t tile_h /*rows*/,
                                            uint32_t row_stride /*elems*/,
                                            uint32_t pad_int_code,
                                            uint32_t pad_amt_code)
{
  const uint64_t ga = (uint64_t)(uintptr_t)gptr;
  v4u g0 = { 1u,                                   // count=1 (valid user D#)
             lds_off,                               // lds_addr (bytes)
             (uint32_t)ga,                          // global_addr[31:0]
             (uint32_t)((ga >> 32) & 0x01ffffffu)   // global_addr[56:32]
               | (2u << 30) };                      // type=2 ("image")
  v8i g1 = {
    (int)((1u << 16)                 // data_size = 2 bytes
        | (1u << 20)                 // pad_enable
        | (pad_int_code << 22)       // pad_interval
        | (pad_amt_code << 25)),     // pad_amount
    (int)((tile_w & 0xffffu) << 16),                                  // tensor_dim0 lo16 @bit48
    (int)(((tile_w >> 16) & 0xffffu) | ((tile_h & 0xffffu) << 16)),   // tensor_dim0 hi / tensor_dim1 lo
    (int)(((tile_h >> 16) & 0xffffu) | ((tile_w & 0xffffu) << 16)),   // tensor_dim1 hi / tile_dim0
    (int)(tile_h & 0xffffu),                                          // tile_dim1 (tile_dim2=0)
    (int)row_stride,                                                  // tensor_dim0_stride[31:0]
    0, 0 };
  v4i gz = { 0, 0, 0, 0 };
#if defined(__clang_major__) && (__clang_major__ >= 23)
  v8i gz8 = { 0, 0, 0, 0, 0, 0, 0, 0 };
  __builtin_amdgcn_tensor_load_to_lds(g0, g1, gz, gz, gz8, 0);
#else
  __builtin_amdgcn_tensor_load_to_lds(g0, g1, gz, gz, 0);
#endif
}

// ---------------------------------------------------------------------------
// GEMM: Y = (X_f32[8192x1024] * W_f32[1024x1024] + bias) * scale  -> bf16 out
// mode 0: out[((b*H+h)*S + s)*DH + d]   (split-heads, for Q and K)
// mode 1: out[((b*H+h)*DH + d)*S + s]   (split-heads transposed, for V)
// Block 256 threads = 8 waves; block tile 64x128, wave tile 16x64.
// W tile staged in LDS as [col][kpair] u32 (col stride 20 dw -> conflict-free,
// lane fragment = two 16B runs 32B apart = 2x ds_load_b128).
// ---------------------------------------------------------------------------
__global__ __launch_bounds__(256) void gemm_qkv_kernel(
    const float* __restrict__ A, const float* __restrict__ W,
    const float* __restrict__ bias, uint16_t* __restrict__ out,
    float scale, int mode)
{
  __shared__ __align__(16) uint16_t sA[64 * 40];   // padded stride 40 (20 dw)
  __shared__ __align__(16) uint32_t sW[128 * 20];  // [col][kpair] u32 pairs

  const int t    = threadIdx.x;
  const int lane = t & 31;
  const int wave = t >> 5;
  const int l15  = lane & 15;
  const int hsel = (lane >> 4) & 1;

  const int m0 = blockIdx.x * 64;
  const int n0 = blockIdx.y * 128;
  const int waveM = (wave >> 1) * 16;
  const int waveN = (wave & 1) * 64;

  v8f acc[4] = {};

  for (int k0 = 0; k0 < DMODEL; k0 += 32) {
    // stage A tile 64x32 (f32 -> bf16)
    {
      const int row = t >> 2;
      const int cg  = (t & 3) * 8;
      const float* ap = A + (size_t)(m0 + row) * DMODEL + k0 + cg;
      float4 a0 = *(const float4*)ap;
      float4 a1 = *(const float4*)(ap + 4);
      if (k0 + 32 < DMODEL) __builtin_prefetch(ap + 32, 0, 1);
      uint4 pk;
      pk.x = pack2bf(a0.x, a0.y);
      pk.y = pack2bf(a0.z, a0.w);
      pk.z = pack2bf(a1.x, a1.y);
      pk.w = pack2bf(a1.z, a1.w);
      *(uint4*)((char*)sA + row * 80 + cg * 2) = pk;
    }
    // stage W tile 32x128 as (k even,k odd) u32 pairs, k-pairs contiguous per column
    {
      const int kp = t & 15;          // k-pair row 0..15
      const int cg = (t >> 4) * 8;    // col group 0..120
      const float* w0 = W + (size_t)(k0 + 2 * kp) * DMODEL + n0 + cg;
      const float* w1 = w0 + DMODEL;
      float4 w0a = *(const float4*)w0, w0b = *(const float4*)(w0 + 4);
      float4 w1a = *(const float4*)w1, w1b = *(const float4*)(w1 + 4);
      if (k0 + 32 < DMODEL) __builtin_prefetch(w0 + 32 * DMODEL, 0, 1);
      sW[(cg + 0) * 20 + kp] = pack2bf(w0a.x, w1a.x);
      sW[(cg + 1) * 20 + kp] = pack2bf(w0a.y, w1a.y);
      sW[(cg + 2) * 20 + kp] = pack2bf(w0a.z, w1a.z);
      sW[(cg + 3) * 20 + kp] = pack2bf(w0a.w, w1a.w);
      sW[(cg + 4) * 20 + kp] = pack2bf(w0b.x, w1b.x);
      sW[(cg + 5) * 20 + kp] = pack2bf(w0b.y, w1b.y);
      sW[(cg + 6) * 20 + kp] = pack2bf(w0b.z, w1b.z);
      sW[(cg + 7) * 20 + kp] = pack2bf(w0b.w, w1b.w);
    }
    __syncthreads();

    const v16bf af = lds_fragA(sA + (waveM + l15) * 40, hsel * 16);
    #pragma unroll
    for (int j = 0; j < 4; ++j) {
      FragU bf_;
      const uint32_t* colp = sW + (waveN + j * 16 + l15) * 20 + hsel * 4;
      bf_.q[0] = *(const uint4*)(colp);
      bf_.q[1] = *(const uint4*)(colp + 8);
      acc[j] = WMMA_BF16(af, bf_.v, acc[j]);
    }
    __syncthreads();
  }

  #pragma unroll
  for (int j = 0; j < 4; ++j) {
    const int ncol = n0 + waveN + j * 16 + l15;
    const float bv = bias[ncol];
    #pragma unroll
    for (int r = 0; r < 8; ++r) {
      const int m  = m0 + waveM + hsel * 8 + r;
      const float val = (acc[j][r] + bv) * scale;
      const int bb = m >> 10, ss = m & 1023;
      const int hh = ncol >> 6, dd = ncol & 63;
      int idx;
      if (mode == 0) idx = ((bb * NH + hh) * S_LEN + ss) * DH + dd;
      else           idx = ((bb * NH + hh) * DH + dd) * S_LEN + ss;
      out[idx] = f2bf(val);
    }
  }
}

// ---------------------------------------------------------------------------
// Final projection: out_f32 = ctx_bf16[8192x1024] * Wo_f32 + bo
// A tile (already bf16) is staged by the Tensor Data Mover (wave 0 issues,
// TDM row-pad 64B data + 16B -> the 80B conflict-free LDS stride directly).
// ---------------------------------------------------------------------------
__global__ __launch_bounds__(256) void gemm_out_kernel(
    const uint16_t* __restrict__ A, const float* __restrict__ W,
    const float* __restrict__ bias, float* __restrict__ out)
{
  __shared__ __align__(16) uint16_t sA[64 * 40];
  __shared__ __align__(16) uint32_t sW[128 * 20];

  const int t    = threadIdx.x;
  const int lane = t & 31;
  const int wave = t >> 5;
  const int l15  = lane & 15;
  const int hsel = (lane >> 4) & 1;

  const int m0 = blockIdx.x * 64;
  const int n0 = blockIdx.y * 128;
  const int waveM = (wave >> 1) * 16;
  const int waveN = (wave & 1) * 64;

  const uint32_t sA_lds = (uint32_t)(uintptr_t)sA;

  v8f acc[4] = {};

  for (int k0 = 0; k0 < DMODEL; k0 += 32) {
    // A tile 64x32 bf16 via TDM: row 64B (16dw) -> pad_interval=3, +16B pad
    if (wave == 0)
      tdm_load_2d(sA_lds, A + (size_t)m0 * DMODEL + k0, 32u, 64u,
                  (uint32_t)DMODEL, 3u, 3u);
    // stage W tile 32x128 as u32 K-pairs (f32 -> bf16), k-pairs contiguous per column
    {
      const int kp = t & 15;
      const int cg = (t >> 4) * 8;
      const float* w0 = W + (size_t)(k0 + 2 * kp) * DMODEL + n0 + cg;
      const float* w1 = w0 + DMODEL;
      float4 w0a = *(const float4*)w0, w0b = *(const float4*)(w0 + 4);
      float4 w1a = *(const float4*)w1, w1b = *(const float4*)(w1 + 4);
      if (k0 + 32 < DMODEL) __builtin_prefetch(w0 + 32 * DMODEL, 0, 1);
      sW[(cg + 0) * 20 + kp] = pack2bf(w0a.x, w1a.x);
      sW[(cg + 1) * 20 + kp] = pack2bf(w0a.y, w1a.y);
      sW[(cg + 2) * 20 + kp] = pack2bf(w0a.z, w1a.z);
      sW[(cg + 3) * 20 + kp] = pack2bf(w0a.w, w1a.w);
      sW[(cg + 4) * 20 + kp] = pack2bf(w0b.x, w1b.x);
      sW[(cg + 5) * 20 + kp] = pack2bf(w0b.y, w1b.y);
      sW[(cg + 6) * 20 + kp] = pack2bf(w0b.z, w1b.z);
      sW[(cg + 7) * 20 + kp] = pack2bf(w0b.w, w1b.w);
    }
    if (wave == 0) __builtin_amdgcn_s_wait_tensorcnt((short)0);
    __syncthreads();

    const v16bf af = lds_fragA(sA + (waveM + l15) * 40, hsel * 16);
    #pragma unroll
    for (int j = 0; j < 4; ++j) {
      FragU bf_;
      const uint32_t* colp = sW + (waveN + j * 16 + l15) * 20 + hsel * 4;
      bf_.q[0] = *(const uint4*)(colp);
      bf_.q[1] = *(const uint4*)(colp + 8);
      acc[j] = WMMA_BF16(af, bf_.v, acc[j]);
    }
    __syncthreads();
  }

  #pragma unroll
  for (int j = 0; j < 4; ++j) {
    const int ncol = n0 + waveN + j * 16 + l15;
    const float bv = bias[ncol];
    #pragma unroll
    for (int r = 0; r < 8; ++r) {
      const int m = m0 + waveM + hsel * 8 + r;
      out[(size_t)m * DMODEL + ncol] = acc[j][r] + bv;
    }
  }
}

// ---------------------------------------------------------------------------
// Flash attention over one (b,h). Block = 128 threads = 4 waves; each wave owns
// a 16-query strip. K / V^T 32-key tiles are DMA'd into double-buffered LDS by
// the Tensor Data Mover (issued by wave 0, tracked on TENSORcnt, overlapped
// with compute), with TDM row padding producing the bank-conflict-free padded
// strides directly. scores^T (keys x queries) keeps softmax stats per-lane
// (one SWAPX16 swizzle to merge halves); P is packed C->B within each lane,
// no lane exchange; ctx^T = V^T x P via WMMA with online-softmax rescaling.
// ---------------------------------------------------------------------------
__global__ __launch_bounds__(128) void attn_kernel(
    const uint16_t* __restrict__ qg, const uint16_t* __restrict__ kg,
    const uint16_t* __restrict__ vtg, const unsigned char* __restrict__ maskp,
    uint16_t* __restrict__ ctx)
{
  constexpr int KELEM = 32 * 72;   // 32 keys x 64 dims, padded stride 72 (144B)
  constexpr int VELEM = 64 * 40;   // 64 dims x 32 keys, padded stride 40 (80B)
  __shared__ __align__(16) uint16_t smem[2 * (KELEM + VELEM)];

  const int t    = threadIdx.x;
  const int lane = t & 31;
  const int wave = t >> 5;
  const int l15  = lane & 15;
  const int hsel = (lane >> 4) & 1;

  const int bh = blockIdx.y;      // b*H + h
  const int b  = bh >> 4;
  const int h  = bh & 15;
  const int q0 = blockIdx.x * 64 + wave * 16;

  const uint32_t smem_base = (uint32_t)(uintptr_t)smem;

  auto issue_tile = [&](int buf, int kb) {
    const uint32_t koff = smem_base + (uint32_t)(buf * (KELEM + VELEM)) * 2u;
    const uint32_t voff = koff + (uint32_t)KELEM * 2u;
    // K tile: 32 rows x 64 bf16 (128B row -> pad_interval=4 (32dw)), +16B pad
    tdm_load_2d(koff, kg + ((size_t)bh * S_LEN + kb) * DH, 64u, 32u, 64u, 4u, 3u);
    // V^T tile: 64 rows x 32 bf16 (64B row -> pad_interval=3 (16dw)), +16B pad
    tdm_load_2d(voff, vtg + (size_t)bh * DH * S_LEN + kb, 32u, 64u, (uint32_t)S_LEN, 3u, 3u);
  };

  // preload Q^T B-fragments: 2 chunks of 32 dims (dims are the K axis)
  FragU qf[2];
  #pragma unroll
  for (int c = 0; c < 2; ++c) {
    #pragma unroll
    for (int v = 0; v < 8; ++v) {
      const int dim = c * 32 + hsel * 8 + (v >> 2) * 16 + 2 * (v & 3);
      qf[c].u[v] = *(const uint32_t*)(qg + ((size_t)bh * S_LEN + q0 + l15) * DH + dim);
    }
  }

  if (wave == 0) issue_tile(0, 0);

  v8f o[4] = {};          // ctx^T accumulators: 4 x (16 dims x 16 queries)
  float mrow = -1e30f;    // running max   (lane n and n^16 stay in sync)
  float lrow = 0.f;       // running sum   (per-lane partial, merged at end)

  for (int it = 0; it < S_LEN / 32; ++it) {
    const int kb  = it * 32;
    const int cur = it & 1;
    if (wave == 0) {
      if (kb + 32 < S_LEN) {
        issue_tile(cur ^ 1, kb + 32);                 // overlap next DMA
        __builtin_amdgcn_s_wait_tensorcnt((short)2);  // current tile (2 ops) done
      } else {
        __builtin_amdgcn_s_wait_tensorcnt((short)0);
      }
    }
    __syncthreads();      // publish TDM-written LDS to all waves

    const uint16_t* sK = smem + cur * (KELEM + VELEM);
    const uint16_t* sV = sK + KELEM;

    // scores^T: two 16(key) x 16(query) tiles, contraction over 64 dims
    v8f c0 = {}, c1 = {};
    #pragma unroll
    for (int c = 0; c < 2; ++c) {
      const v16bf a0 = lds_fragA(sK + l15 * 72,        (c * 32 + hsel * 8) * 2);
      c0 = WMMA_BF16(a0, qf[c].v, c0);
      const v16bf a1 = lds_fragA(sK + (16 + l15) * 72, (c * 32 + hsel * 8) * 2);
      c1 = WMMA_BF16(a1, qf[c].v, c1);
    }

    // mask: 8 consecutive key-bytes per lane per tile (reference: True -> -1e4)
    const unsigned char* mp =
        maskp + ((size_t)b * S_LEN + q0 + l15) * S_LEN + kb + hsel * 8;
    const uint32_t ma = *(const uint32_t*)(mp);
    const uint32_t mb = *(const uint32_t*)(mp + 4);
    const uint32_t mc = *(const uint32_t*)(mp + 16);
    const uint32_t md = *(const uint32_t*)(mp + 20);

    float s0[8], s1[8];
    #pragma unroll
    for (int r = 0; r < 8; ++r) {
      const uint32_t b0 = ((r < 4 ? ma : mb) >> ((r & 3) * 8)) & 0xffu;
      const uint32_t b1 = ((r < 4 ? mc : md) >> ((r & 3) * 8)) & 0xffu;
      s0[r] = b0 ? -10000.f : c0[r];
      s1[r] = b1 ? -10000.f : c1[r];
    }

    // online softmax
    float tmax = s0[0];
    #pragma unroll
    for (int r = 1; r < 8; ++r) tmax = fmaxf(tmax, s0[r]);
    #pragma unroll
    for (int r = 0; r < 8; ++r) tmax = fmaxf(tmax, s1[r]);
    tmax = fmaxf(tmax, swz16(tmax));          // merge lane<->lane^16 halves

    const float mnew = fmaxf(mrow, tmax);
    const float corr = __expf(mrow - mnew);
    mrow = mnew;

    float p0[8], p1[8], psum = 0.f;
    #pragma unroll
    for (int r = 0; r < 8; ++r) { p0[r] = __expf(s0[r] - mnew); psum += p0[r]; }
    #pragma unroll
    for (int r = 0; r < 8; ++r) { p1[r] = __expf(s1[r] - mnew); psum += p1[r]; }
    lrow = lrow * corr + psum;

    #pragma unroll
    for (int j = 0; j < 4; ++j)
      #pragma unroll
      for (int r = 0; r < 8; ++r) o[j][r] *= corr;

    // rearrange P (C layout, keys x queries) into B layout (32 keys x 16 q)
    FragU bp;
    #pragma unroll
    for (int v = 0; v < 4; ++v) {
      const float lo_o = swz16(p1[2 * v]);
      const float hi_o = swz16(p1[2 * v + 1]);
      const float lo = p0[2 * v];
      const float hi = p0[2 * v + 1];
      bp.u[v] = pack2bf(lo, hi);
    }
    #pragma unroll
    for (int v = 4; v < 8; ++v) {
      const float lo_o = swz16(p0[2 * v - 8]);
      const float hi_o = swz16(p0[2 * v - 7]);
      const float lo = p1[2 * v - 8];
      const float hi = p1[2 * v - 7];
      bp.u[v] = pack2bf(lo, hi);
    }

    // ctx^T += V^T(16 dims x 32 keys) x P(32 keys x 16 queries)
    #pragma unroll
    for (int j = 0; j < 4; ++j) {
      const v16bf av = lds_fragA(sV + (j * 16 + l15) * 40, hsel * 16);
      o[j] = WMMA_BF16(av, bp.v, o[j]);
    }
    __syncthreads();      // all waves done reading this buffer
  }

  const float lfull = lrow + swz16(lrow);
  const float inv = 1.f / lfull;

  // each lane's 8 dims per accumulator are contiguous -> one b128 store each
  #pragma unroll
  for (int j = 0; j < 4; ++j) {
    uint4 st;
    st.x = pack2bf(o[j][0] * inv, o[j][1] * inv);
    st.y = pack2bf(o[j][2] * inv, o[j][3] * inv);
    st.z = pack2bf(o[j][4] * inv, o[j][5] * inv);
    st.w = pack2bf(o[j][6] * inv, o[j][7] * inv);
    const int dim = j * 16 + hsel * 8;
    *(uint4*)(ctx + ((size_t)b * S_LEN + q0 + l15) * DMODEL + h * DH + dim) = st;
  }
}

// ---------------------------------------------------------------------------
extern "C" void kernel_launch(void* const* d_in, const int* in_sizes, int n_in,
                              void* d_out, int out_size, void* d_ws, size_t ws_size,
                              hipStream_t stream) {
  (void)in_sizes; (void)n_in; (void)out_size; (void)ws_size;

  const float* query = (const float*)d_in[0];
  const float* key_  = (const float*)d_in[1];
  const float* value = (const float*)d_in[2];
  const unsigned char* mask = (const unsigned char*)d_in[3];
  const float* Wq = (const float*)d_in[4];
  const float* bq = (const float*)d_in[5];
  const float* Wk = (const float*)d_in[6];
  const float* bk = (const float*)d_in[7];
  const float* Wv = (const float*)d_in[8];
  const float* bv = (const float*)d_in[9];
  const float* Wo = (const float*)d_in[10];
  const float* bo = (const float*)d_in[11];

  const size_t per = (size_t)BATCH * NH * S_LEN * DH;  // 8,388,608 elements
  uint16_t* qw  = (uint16_t*)d_ws;
  uint16_t* kw  = qw + per;
  uint16_t* vtw = kw + per;
  uint16_t* cw  = vtw + per;

  dim3 gG(BATCH * S_LEN / 64, DMODEL / 128);  // (128, 8)
  gemm_qkv_kernel<<<gG, 256, 0, stream>>>(query, Wq, bq, qw, 0.125f, 0); // Q pre-scaled 1/sqrt(64)
  gemm_qkv_kernel<<<gG, 256, 0, stream>>>(key_,  Wk, bk, kw, 1.0f,  0);
  gemm_qkv_kernel<<<gG, 256, 0, stream>>>(value, Wv, bv, vtw, 1.0f, 1);  // V stored transposed

  attn_kernel<<<dim3(S_LEN / 64, BATCH * NH), 128, 0, stream>>>(qw, kw, vtw, mask, cw);

  gemm_out_kernel<<<gG, 256, 0, stream>>>(cw, Wo, bo, (float*)d_out);
}
